// AdaptiveJacobianPrunedViT_13134009991251
// MI455X (gfx1250) — matrix operand order repaired and checked by reading, not compiled, not run on hardware
//
#include <hip/hip_runtime.h>

// ---------------------------------------------------------------------------
// Types for CDNA5 WMMA
// ---------------------------------------------------------------------------
typedef __attribute__((ext_vector_type(16))) __bf16 v16bf;
typedef __attribute__((ext_vector_type(8)))  float  v8f;

union FragB { v16bf v; unsigned int u[8]; };
union FragC { v8f  v; float f[8]; };
union LdU   { uint2 v; unsigned short s[4]; };

#define DEVFN __device__ __forceinline__

DEVFN unsigned short f2bf(float x) {
    unsigned int u = __float_as_uint(x);
    unsigned int r = u + 0x7FFFu + ((u >> 16) & 1u);   // round-to-nearest-even
    return (unsigned short)(r >> 16);
}
DEVFN float bf2f(unsigned short h) { return __uint_as_float(((unsigned int)h) << 16); }

// CDNA5 async global->LDS copy (ASYNCcnt-tracked, ISA 15.18 op 97)
DEVFN void async_g2l_b64(unsigned ldsAddr, const unsigned short* gaddr) {
    asm volatile("global_load_async_to_lds_b64 %0, %1, off"
                 :: "v"(ldsAddr), "v"(gaddr) : "memory");
}
DEVFN void wait_async0() {
    asm volatile("s_wait_asynccnt 0" ::: "memory");
}

// ---------------------------------------------------------------------------
// Problem constants (ViT-Base, B=32)
// ---------------------------------------------------------------------------
constexpr int Bsz  = 32;
constexpr int Dm   = 768;
constexpr int FF   = 3072;
constexpr int NP   = 196;      // patches
constexpr int Hh   = 12;       // heads
constexpr int DH   = 64;       // head dim
constexpr int Tmax = 224;      // padded token count (197 -> 224, mult of 32)
constexpr int T16  = Tmax / 16;
constexpr int NTOT = Bsz * Tmax;

// ---------------------------------------------------------------------------
// Generic bf16 WMMA GEMM:  out = act(A[M,K] @ W[K,Nld] + bias) (+resid)
// W is padded so Nld % 128 == 0; epilogue clips to Nout.
// Row mapping: row r -> (b = r/ntok, t = r%ntok), buffer offset (b*tmax+t)
// ---------------------------------------------------------------------------
__global__ __launch_bounds__(256)
void k_gemm(const unsigned short* __restrict__ A, const unsigned short* __restrict__ W,
            const float* __restrict__ bias, const float* __restrict__ resid,
            float* __restrict__ out32, unsigned short* __restrict__ out16,
            int K, int Nld, int Nout, int ldA, int ldC,
            int Bc, int tmax, int ntokFixed, const int* __restrict__ ntokPtr, int ntokAdd,
            int gelu)
{
    __shared__ __align__(16) unsigned short As[64][36];
    __shared__ __align__(16) unsigned short BsT[128][36];

    int ntok = (ntokPtr ? *ntokPtr : ntokFixed) + ntokAdd;
    int rows = Bc * ntok;
    int rowBase = blockIdx.y * 64;
    if (rowBase >= rows) return;
    int colBase = blockIdx.x * 128;
    int tid  = threadIdx.x;
    int lane = tid & 31, wave = tid >> 5;
    int wm = wave >> 2, wn = wave & 3;

    // ---- hoisted staging descriptors (fixed per thread across K steps) ----
    const unsigned short* aPtr[2];
    unsigned aLds[2];
#pragma unroll
    for (int it = 0; it < 2; ++it) {
        int idx = tid + it * 256;              // 0..511 -> 64 rows x 8 chunks
        int r = idx >> 3, c4 = (idx & 7) * 4;
        int gr = rowBase + r;
        int grc = (gr < rows) ? gr : 0;        // clamp: garbage rows never stored
        int b = grc / ntok, tt = grc - b * ntok;
        aPtr[it] = A + ((size_t)(b * tmax + tt)) * ldA + c4;
        aLds[it] = (unsigned)(unsigned long long)&As[r][c4];
    }
    const unsigned short* bPtr[4];
    unsigned short* bDst[4];
#pragma unroll
    for (int it = 0; it < 4; ++it) {
        int idx = tid + it * 256;              // 0..1023 -> 32 rows x 32 chunks
        int r = idx >> 5, c4 = (idx & 31) * 4;
        bPtr[it] = W + (size_t)r * Nld + colBase + c4;
        bDst[it] = &BsT[c4][r];
    }

    FragC acc[2][2];
#pragma unroll
    for (int i = 0; i < 2; ++i)
#pragma unroll
        for (int j = 0; j < 2; ++j)
#pragma unroll
            for (int v = 0; v < 8; ++v) acc[i][j].f[v] = 0.f;

    for (int k0 = 0; k0 < K; k0 += 32) {
        // A tile 64x32: async global->LDS (row-major, no transpose needed)
#pragma unroll
        for (int it = 0; it < 2; ++it) {
            async_g2l_b64(aLds[it], aPtr[it] + k0);
            __builtin_prefetch((const char*)(aPtr[it] + k0) + 64, 0, 1);
        }
        // B tile 32x128 -> BsT[n][k] (transpose scatter)
#pragma unroll
        for (int it = 0; it < 4; ++it) {
            LdU t;
            t.v = *(const uint2*)(bPtr[it] + (size_t)k0 * Nld);
            unsigned short* d = bDst[it];
            d[0] = t.s[0]; d[36] = t.s[1]; d[72] = t.s[2]; d[108] = t.s[3];
        }
        wait_async0();
        __syncthreads();

        FragB fa[2], fb[2];
#pragma unroll
        for (int i = 0; i < 2; ++i) {
            int row = wm * 32 + i * 16 + (lane & 15);
            int kg  = (lane >> 4) * 16;
            const unsigned short* rp = &As[row][0];
#pragma unroll
            for (int v = 0; v < 8; ++v) {
                int kk = kg + 2 * v;
                fa[i].u[v] = *(const unsigned int*)(rp + kk);
            }
        }
#pragma unroll
        for (int j = 0; j < 2; ++j) {
            int col = wn * 32 + j * 16 + (lane & 15);
            int kb  = (lane >> 4) * 16;
            const unsigned short* cp = &BsT[col][kb];
#pragma unroll
            for (int v = 0; v < 8; ++v) fb[j].u[v] = *(const unsigned int*)(cp + 2 * v);
        }
#pragma unroll
        for (int i = 0; i < 2; ++i)
#pragma unroll
            for (int j = 0; j < 2; ++j)
                acc[i][j].v = __builtin_amdgcn_wmma_f32_16x16x32_bf16(
                    false, fa[i].v, false, fb[j].v, (short)0, acc[i][j].v, false, false);
        __syncthreads();
    }

    // epilogue
#pragma unroll
    for (int i = 0; i < 2; ++i) {
#pragma unroll
        for (int j = 0; j < 2; ++j) {
            int gc = colBase + wn * 32 + j * 16 + (lane & 15);
            if (gc >= Nout) continue;
            float bv = bias ? bias[gc] : 0.f;
#pragma unroll
            for (int v = 0; v < 8; ++v) {
                int gr = rowBase + wm * 32 + i * 16 + v + ((lane >> 4) * 8);
                if (gr >= rows) continue;
                int b = gr / ntok, tt = gr - b * ntok;
                size_t co = (size_t)(b * tmax + tt) * ldC + gc;
                float val = acc[i][j].f[v] + bv;
                if (gelu) {
                    float x3 = val * val * val;
                    val = 0.5f * val * (1.f + tanhf(0.7978845608028654f * (val + 0.044715f * x3)));
                }
                if (resid) val += resid[co];
                if (out32) out32[co] = val;
                if (out16) out16[co] = f2bf(val);
            }
        }
    }
}

// ---------------------------------------------------------------------------
// Attention: S = Q K^T * scale, one 16x16 tile per wave (K = 64 -> 2 WMMA)
// q,k: bf16 [B*H, Tmax, 64] ; s: f32 [B*H, Tmax, Tmax]
// ---------------------------------------------------------------------------
__global__ __launch_bounds__(32)
void k_attn_scores(const unsigned short* __restrict__ q, const unsigned short* __restrict__ k,
                   float* __restrict__ s, const int* __restrict__ nPtr)
{
    int T = *nPtr + 1;
    int bh = blockIdx.y;
    int mt = blockIdx.x / T16, nt = blockIdx.x % T16;
    if (mt * 16 >= T || nt * 16 >= T) return;
    int lane = threadIdx.x;
    FragC acc;
#pragma unroll
    for (int v = 0; v < 8; ++v) acc.f[v] = 0.f;
    const unsigned short* qb = q + (size_t)bh * Tmax * DH;
    const unsigned short* kb = k + (size_t)bh * Tmax * DH;
#pragma unroll
    for (int ks = 0; ks < 2; ++ks) {
        FragB fa, fb;
        {
            int row = mt * 16 + (lane & 15);
            int kg  = (lane >> 4) * 16 + ks * 32;
            const unsigned short* rp = qb + (size_t)row * DH;
#pragma unroll
            for (int v = 0; v < 8; ++v) {
                int kk = kg + 2 * v;
                fa.u[v] = *(const unsigned int*)(rp + kk);
            }
        }
        {
            int col = nt * 16 + (lane & 15);
            int kbb = (lane >> 4) * 16 + ks * 32;
            const unsigned short* cp = kb + (size_t)col * DH + kbb;
#pragma unroll
            for (int v = 0; v < 8; ++v) fb.u[v] = *(const unsigned int*)(cp + 2 * v);
        }
        acc.v = __builtin_amdgcn_wmma_f32_16x16x32_bf16(false, fa.v, false, fb.v,
                                                        (short)0, acc.v, false, false);
    }
#pragma unroll
    for (int v = 0; v < 8; ++v) {
        int row = mt * 16 + v + ((lane >> 4) * 8);
        int col = nt * 16 + (lane & 15);
        s[((size_t)bh * Tmax + row) * Tmax + col] = acc.f[v] * 0.125f;
    }
}

// row softmax: f32 scores -> bf16 probs (padded cols zeroed)
__global__ __launch_bounds__(32)
void k_softmax(const float* __restrict__ s, unsigned short* __restrict__ p,
               const int* __restrict__ nPtr)
{
    int T = *nPtr + 1;
    int row = blockIdx.x % Tmax, bh = blockIdx.x / Tmax;
    const float* sr = s + ((size_t)bh * Tmax + row) * Tmax;
    unsigned short* pr = p + ((size_t)bh * Tmax + row) * Tmax;
    int lane = threadIdx.x;
    if (row >= T) { for (int j = lane; j < Tmax; j += 32) pr[j] = 0; return; }
    float m = -1e30f;
    for (int j = lane; j < T; j += 32) m = fmaxf(m, sr[j]);
    for (int o = 16; o > 0; o >>= 1) m = fmaxf(m, __shfl_xor(m, o, 32));
    float sum = 0.f;
    for (int j = lane; j < T; j += 32) sum += __expf(sr[j] - m);
    for (int o = 16; o > 0; o >>= 1) sum += __shfl_xor(sum, o, 32);
    float inv = 1.f / sum;
    for (int j = lane; j < Tmax; j += 32)
        pr[j] = (j < T) ? f2bf(__expf(sr[j] - m) * inv) : (unsigned short)0;
}

// O = P V : probs bf16 [B*H,Tmax,Tmax] x vT bf16 [B*H,64,Tmax] -> o bf16 [B,Tmax,768]
__global__ __launch_bounds__(32)
void k_attn_out(const unsigned short* __restrict__ p, const unsigned short* __restrict__ vt,
                unsigned short* __restrict__ o, const int* __restrict__ nPtr)
{
    int T = *nPtr + 1;
    int bh = blockIdx.y;
    int mt = blockIdx.x >> 2, dt = blockIdx.x & 3;
    if (mt * 16 >= T) return;
    int lane = threadIdx.x;
    int b = bh / Hh, h = bh - b * Hh;
    FragC acc;
#pragma unroll
    for (int v = 0; v < 8; ++v) acc.f[v] = 0.f;
    const unsigned short* pb = p + (size_t)bh * Tmax * Tmax;
    const unsigned short* vb = vt + (size_t)bh * DH * Tmax;
#pragma unroll
    for (int ks = 0; ks < Tmax / 32; ++ks) {
        FragB fa, fb;
        {
            int row = mt * 16 + (lane & 15);
            int kg  = (lane >> 4) * 16 + ks * 32;
            const unsigned short* rp = pb + (size_t)row * Tmax;
#pragma unroll
            for (int v = 0; v < 8; ++v) {
                int kk = kg + 2 * v;
                fa.u[v] = *(const unsigned int*)(rp + kk);
            }
        }
        {
            int col = dt * 16 + (lane & 15);
            int kbb = (lane >> 4) * 16 + ks * 32;
            const unsigned short* cp = vb + (size_t)col * Tmax + kbb;
#pragma unroll
            for (int v = 0; v < 8; ++v) fb.u[v] = *(const unsigned int*)(cp + 2 * v);
        }
        acc.v = __builtin_amdgcn_wmma_f32_16x16x32_bf16(false, fa.v, false, fb.v,
                                                        (short)0, acc.v, false, false);
    }
#pragma unroll
    for (int v = 0; v < 8; ++v) {
        int row = mt * 16 + v + ((lane >> 4) * 8);
        if (row < T)
            o[((size_t)(b * Tmax + row)) * Dm + h * DH + dt * 16 + (lane & 15)] = f2bf(acc.f[v]);
    }
}

// ---------------------------------------------------------------------------
// LayerNorm: fp32 row -> bf16 row (same [b*Tmax+t] mapping)
// ---------------------------------------------------------------------------
__global__ __launch_bounds__(256)
void k_layernorm(const float* __restrict__ x, const float* __restrict__ w,
                 const float* __restrict__ bb, unsigned short* __restrict__ y,
                 const int* __restrict__ nPtr, int ntokAdd)
{
    __shared__ float red[256];
    int ntok = *nPtr + ntokAdd;
    int t = blockIdx.x % Tmax, b = blockIdx.x / Tmax;
    if (t >= ntok) return;
    const float* xr = x + ((size_t)(b * Tmax + t)) * Dm;
    int tid = threadIdx.x;
    float s = 0.f;
    for (int i = tid; i < Dm; i += 256) s += xr[i];
    red[tid] = s; __syncthreads();
    for (int o = 128; o > 0; o >>= 1) { if (tid < o) red[tid] += red[tid + o]; __syncthreads(); }
    float mu = red[0] / Dm; __syncthreads();
    float v = 0.f;
    for (int i = tid; i < Dm; i += 256) { float d = xr[i] - mu; v += d * d; }
    red[tid] = v; __syncthreads();
    for (int o = 128; o > 0; o >>= 1) { if (tid < o) red[tid] += red[tid + o]; __syncthreads(); }
    float rs = rsqrtf(red[0] / Dm + 1e-6f);
    unsigned short* yr = y + ((size_t)(b * Tmax + t)) * Dm;
    for (int i = tid; i < Dm; i += 256) yr[i] = f2bf((xr[i] - mu) * rs * w[i] + bb[i]);
}

// final LN of CLS rows -> bf16 [32,768] contiguous
__global__ __launch_bounds__(256)
void k_ln_cls(const float* __restrict__ x, const float* __restrict__ w,
              const float* __restrict__ bb, unsigned short* __restrict__ y)
{
    __shared__ float red[256];
    int b = blockIdx.x;
    const float* xr = x + (size_t)(b * Tmax) * Dm;
    int tid = threadIdx.x;
    float s = 0.f;
    for (int i = tid; i < Dm; i += 256) s += xr[i];
    red[tid] = s; __syncthreads();
    for (int o = 128; o > 0; o >>= 1) { if (tid < o) red[tid] += red[tid + o]; __syncthreads(); }
    float mu = red[0] / Dm; __syncthreads();
    float v = 0.f;
    for (int i = tid; i < Dm; i += 256) { float d = xr[i] - mu; v += d * d; }
    red[tid] = v; __syncthreads();
    for (int o = 128; o > 0; o >>= 1) { if (tid < o) red[tid] += red[tid + o]; __syncthreads(); }
    float rs = rsqrtf(red[0] / Dm + 1e-6f);
    for (int i = tid; i < Dm; i += 256) y[(size_t)b * Dm + i] = f2bf((xr[i] - mu) * rs * w[i] + bb[i]);
}

// ---------------------------------------------------------------------------
// qkv split: [b*Tmax+t][2304] -> q,k [B*H,Tmax,64], vT [B*H,64,Tmax] (pad-zeroed)
// ---------------------------------------------------------------------------
__global__ __launch_bounds__(256)
void k_split_qkv(const unsigned short* __restrict__ qkv, unsigned short* __restrict__ q,
                 unsigned short* __restrict__ kk, unsigned short* __restrict__ vt,
                 const int* __restrict__ nPtr)
{
    int T = *nPtr + 1;
    int t = blockIdx.x % Tmax, b = blockIdx.x / Tmax;
    int tid = threadIdx.x;
    bool live = (t < T);
    const unsigned short* row = qkv + ((size_t)(b * Tmax + t)) * (3 * Dm);
    for (int e = tid; e < 3 * Dm; e += 256) {
        int which = e / Dm; int r = e - which * Dm; int h = r >> 6; int d = r & 63;
        unsigned short val = live ? row[e] : (unsigned short)0;
        size_t bh = (size_t)(b * Hh + h);
        if (which == 0)      q [(bh * Tmax + t) * DH + d] = val;
        else if (which == 1) kk[(bh * Tmax + t) * DH + d] = val;
        else                 vt[(bh * DH + d) * Tmax + t] = val;
    }
}

// ---------------------------------------------------------------------------
// Cheap CLS-row attention scoring, per (b,h) block; deterministic (no atomics)
// ---------------------------------------------------------------------------
__global__ __launch_bounds__(256)
void k_cheap(const unsigned short* __restrict__ q, const unsigned short* __restrict__ kk,
             const unsigned short* __restrict__ vt, float* __restrict__ imp2,
             float* __restrict__ massArr, float* __restrict__ rhoArr,
             const int* __restrict__ nPtr)
{
    __shared__ float qs[DH];
    __shared__ float red[256];
    int bh = blockIdx.x;
    int tid = threadIdx.x;
    int T = *nPtr + 1;
    const unsigned short* qrow = q + (size_t)bh * Tmax * DH;    // t = 0 (CLS)
    if (tid < DH) qs[tid] = bf2f(qrow[tid]);
    __syncthreads();
    int t = tid;
    float logit = -1e30f;
    if (t < T) {
        const unsigned short* kr = kk + ((size_t)bh * Tmax + t) * DH;
        float acc = 0.f;
        for (int d = 0; d < DH; ++d) acc += qs[d] * bf2f(kr[d]);
        logit = acc * 0.125f;
    }
    red[tid] = logit; __syncthreads();
    for (int o = 128; o > 0; o >>= 1) { if (tid < o) red[tid] = fmaxf(red[tid], red[tid + o]); __syncthreads(); }
    float m = red[0]; __syncthreads();
    float e = (t < T) ? __expf(logit - m) : 0.f;
    red[tid] = e; __syncthreads();
    for (int o = 128; o > 0; o >>= 1) { if (tid < o) red[tid] += red[tid + o]; __syncthreads(); }
    float a = e / red[0];
    __syncthreads();

    float av = 0.f;
    if (t >= 1 && t < T) {
        float vn = 0.f;
        for (int d = 0; d < DH; ++d) {
            float x = bf2f(vt[((size_t)bh * DH + d) * Tmax + t]);
            vn += x * x;
        }
        av = a * sqrtf(vn);
    }
    if (t >= 1 && t <= Tmax) imp2[(size_t)bh * Tmax + (t - 1)] = av;

    float massP = (t >= 1 && t < T) ? a : 0.f;
    red[tid] = massP; __syncthreads();
    for (int o = 128; o > 0; o >>= 1) { if (tid < o) red[tid] += red[tid + o]; __syncthreads(); }
    if (tid == 0) massArr[bh] = red[0];
    __syncthreads();
    float entP = (t < T) ? (-a * __logf(a + 1e-6f)) : 0.f;
    red[tid] = entP; __syncthreads();
    for (int o = 128; o > 0; o >>= 1) { if (tid < o) red[tid] += red[tid + o]; __syncthreads(); }
    if (tid == 0) rhoArr[bh] = red[0] / __logf((float)T);
}

// deterministic reduction over 384 (b,h) slots
__global__ __launch_bounds__(32)
void k_reduce_cheap(const float* __restrict__ imp2, const float* __restrict__ massArr,
                    const float* __restrict__ rhoArr, float* __restrict__ imp,
                    float* __restrict__ fmeta)
{
    int lane = threadIdx.x;
    int bid = blockIdx.x;
    float s = 0.f;
    const float inv = 1.f / (float)(Bsz * Hh);
    if (bid < Tmax) {
        for (int bh = lane; bh < Bsz * Hh; bh += 32) s += imp2[(size_t)bh * Tmax + bid];
        for (int o = 16; o > 0; o >>= 1) s += __shfl_xor(s, o, 32);
        if (lane == 0) imp[bid] = s * inv;
    } else if (bid == Tmax) {
        for (int bh = lane; bh < Bsz * Hh; bh += 32) s += massArr[bh];
        for (int o = 16; o > 0; o >>= 1) s += __shfl_xor(s, o, 32);
        if (lane == 0) fmeta[5] = s * inv;
    } else {
        for (int bh = lane; bh < Bsz * Hh; bh += 32) s += rhoArr[bh];
        for (int o = 16; o > 0; o >>= 1) s += __shfl_xor(s, o, 32);
        if (lane == 0) fmeta[6] = s * inv;
    }
}

// meta[0]=N_cur, meta[1]=N_next, fmeta[4]=prev_mass, fmeta[5]=mass, fmeta[6]=rho
__global__ void k_decide(int* meta)
{
    float* fm = (float*)meta;
    int N = meta[0];
    int Nn = N;
    if (N > 16) {
        float mass = fm[5], rho = fm[6], prev = fm[4];
        float keep = 1.f - 0.01f * rho * (prev / (mass + 1e-6f));
        keep = fminf(fmaxf(keep, 0.f), 1.f);
        int cand = (int)((float)N * keep);
        if (cand < 16) cand = 16;
        if (cand < N) Nn = cand;
        fm[4] = mass;
    }
    meta[1] = Nn;
}

// top-k by rank (ties -> lower index, matching lax.top_k), sorted keep list
__global__ __launch_bounds__(256)
void k_rank(const float* __restrict__ imp, const int* __restrict__ meta, int* __restrict__ map)
{
    __shared__ unsigned char keep[256];
    int N = meta[0], Nn = meta[1];
    int tid = threadIdx.x;
    keep[tid] = 0;
    __syncthreads();
    if (Nn >= N) {
        if (tid <= N) map[tid] = tid;      // identity: tokens 0..N
        return;
    }
    for (int i = tid; i < N; i += 256) {
        float vi = imp[i];
        int rank = 0;
        for (int j = 0; j < N; ++j) {
            float vj = imp[j];
            rank += (vj > vi) || (vj == vi && j < i);
        }
        keep[i] = (rank < Nn) ? 1 : 0;
    }
    __syncthreads();
    if (tid == 0) {
        map[0] = 0;
        int c = 1;
        for (int i = 0; i < N; ++i) if (keep[i]) map[c++] = i + 1;
    }
}

__global__ __launch_bounds__(256)
void k_gather(const float* __restrict__ src, float* __restrict__ dst,
              const int* __restrict__ meta, const int* __restrict__ map)
{
    int t = blockIdx.x % Tmax, b = blockIdx.x / Tmax;
    int Tn = meta[1] + 1;
    if (t >= Tn) return;
    int st = map[t];
    const float* sr = src + ((size_t)(b * Tmax + st)) * Dm;
    float* dr = dst + ((size_t)(b * Tmax + t)) * Dm;
    for (int i = threadIdx.x; i < Dm; i += 256) dr[i] = sr[i];
}

__global__ void k_commit(int* meta) { meta[0] = meta[1]; }
__global__ void k_init(int* meta)   { meta[0] = NP; ((float*)meta)[4] = 1.0f; }

// ---------------------------------------------------------------------------
// fp32 -> bf16 weight conversion (plain + column-padded variants)
// ---------------------------------------------------------------------------
__global__ __launch_bounds__(256)
void k_cvt_bf16(const float* __restrict__ s, unsigned short* __restrict__ d, long n)
{
    long i = (long)blockIdx.x * 256 + threadIdx.x;
    long stride = (long)gridDim.x * 256;
    for (; i < n; i += stride) d[i] = f2bf(s[i]);
}

__global__ __launch_bounds__(256)
void k_cvt_pad(const float* __restrict__ s, unsigned short* __restrict__ d,
               int K, int Ns, int Nd)
{
    long total = (long)K * Nd;
    long stride = (long)gridDim.x * 256;
    for (long i = (long)blockIdx.x * 256 + threadIdx.x; i < total; i += stride) {
        int k = (int)(i / Nd), n = (int)(i - (long)k * Nd);
        d[i] = (n < Ns) ? f2bf(s[(long)k * Ns + n]) : (unsigned short)0;
    }
}

// patchify: img [32,3,224,224] -> patches bf16 [32*196, 768] (c*256 + py*16 + px)
__global__ __launch_bounds__(256)
void k_patchify(const float* __restrict__ img, unsigned short* __restrict__ out)
{
    int p = blockIdx.x;
    int b = p / NP, pp = p - b * NP;
    int ph = pp / 14, pw = pp - ph * 14;
    int tid = threadIdx.x;
#pragma unroll
    for (int j = 0; j < 3; ++j) {
        int e = tid + j * 256;
        int c = e >> 8; int rem = e & 255; int py = rem >> 4; int px = rem & 15;
        float v = img[(((size_t)(b * 3 + c)) * 224 + (ph * 16 + py)) * 224 + (pw * 16 + px)];
        out[(size_t)p * Dm + e] = f2bf(v);
    }
}

// xx[b][t] = (t==0 ? cls : patch_out[b][t-1]) + pos[t]
__global__ __launch_bounds__(256)
void k_assemble(const float* __restrict__ pout, const float* __restrict__ cls,
                const float* __restrict__ pos, float* __restrict__ xx)
{
    int g = blockIdx.x;
    int b = g / 197, t = g - b * 197;
    int tid = threadIdx.x;
#pragma unroll
    for (int j = 0; j < 3; ++j) {
        int d = tid + j * 256;
        float v = (t == 0) ? cls[d] : pout[((size_t)(b * NP + (t - 1))) * Dm + d];
        xx[((size_t)(b * Tmax + t)) * Dm + d] = v + pos[(size_t)t * Dm + d];
    }
}

// ---------------------------------------------------------------------------
// Launch
// ---------------------------------------------------------------------------
extern "C" void kernel_launch(void* const* d_in, const int* in_sizes, int n_in,
                              void* d_out, int out_size, void* d_ws, size_t ws_size,
                              hipStream_t stream)
{
    (void)in_sizes; (void)n_in; (void)out_size; (void)ws_size;
    const float* in_x     = (const float*)d_in[0];
    const float* patch_w  = (const float*)d_in[1];
    const float* patch_b  = (const float*)d_in[2];
    const float* cls_tok  = (const float*)d_in[3];
    const float* pos_emb  = (const float*)d_in[4];
    const float* norm1_w  = (const float*)d_in[5];
    const float* norm1_b  = (const float*)d_in[6];
    const float* qkv_w    = (const float*)d_in[7];
    const float* qkv_b    = (const float*)d_in[8];
    const float* proj_w   = (const float*)d_in[9];
    const float* proj_b   = (const float*)d_in[10];
    const float* norm2_w  = (const float*)d_in[11];
    const float* norm2_b  = (const float*)d_in[12];
    const float* fc1_w    = (const float*)d_in[13];
    const float* fc1_b    = (const float*)d_in[14];
    const float* fc2_w    = (const float*)d_in[15];
    const float* fc2_b    = (const float*)d_in[16];
    const float* norm_w   = (const float*)d_in[17];
    const float* norm_b   = (const float*)d_in[18];
    const float* head_w   = (const float*)d_in[19];
    const float* head_b   = (const float*)d_in[20];

    char* ws = (char*)d_ws;
    size_t off = 0;
    auto take = [&](size_t bytes) { size_t o = off; off += (bytes + 255) & ~(size_t)255; return o; };

    unsigned short* wqkv   = (unsigned short*)(ws + take(12UL * 768 * 2304 * 2));
    unsigned short* wproj  = (unsigned short*)(ws + take(12UL * 768 * 768 * 2));
    unsigned short* wf1    = (unsigned short*)(ws + take(12UL * 768 * 3072 * 2));
    unsigned short* wf2    = (unsigned short*)(ws + take(12UL * 3072 * 768 * 2));
    unsigned short* wpatch = (unsigned short*)(ws + take(768UL * 768 * 2));
    unsigned short* whead  = (unsigned short*)(ws + take(768UL * 1024 * 2));   // padded to 1024
    float*          xxA    = (float*)(ws + take((size_t)NTOT * Dm * 4));
    float*          xxB    = (float*)(ws + take((size_t)NTOT * Dm * 4));
    unsigned short* xn     = (unsigned short*)(ws + take((size_t)NTOT * Dm * 2));
    unsigned short* xn2    = (unsigned short*)(ws + take((size_t)NTOT * Dm * 2));
    unsigned short* qkvb   = (unsigned short*)(ws + take((size_t)NTOT * 3 * Dm * 2));
    unsigned short* qh     = (unsigned short*)(ws + take((size_t)Bsz * Hh * Tmax * DH * 2));
    unsigned short* kh     = (unsigned short*)(ws + take((size_t)Bsz * Hh * Tmax * DH * 2));
    unsigned short* vth    = (unsigned short*)(ws + take((size_t)Bsz * Hh * DH * Tmax * 2));
    unsigned short* probs  = (unsigned short*)(ws + take((size_t)Bsz * Hh * Tmax * Tmax * 2));
    unsigned short* obuf   = (unsigned short*)(ws + take((size_t)NTOT * Dm * 2));
    unsigned short* clsb   = (unsigned short*)(ws + take((size_t)Bsz * Dm * 2));
    int*            meta   = (int*)(ws + take(4096));
    float*          imp2   = (float*)(ws + take(((size_t)Bsz * Hh * Tmax + 1024) * 4));
    // big region (time-aliased): scores f32, patch staging, MLP hidden
    char* big = ws + take((size_t)Bsz * Hh * Tmax * Tmax * 4);
    float*          scores  = (float*)big;
    unsigned short* hbuf    = (unsigned short*)big;                        // after scores dead
    unsigned short* patches = (unsigned short*)big;                        // prologue only
    float*          pout    = (float*)(big + (((size_t)Bsz * NP * Dm * 2 + 255) & ~(size_t)255));

    float* fmeta   = (float*)meta;
    float* imp     = fmeta + 16;       // 224 floats
    int*   map     = meta + 256;       // 256 ints
    float* massArr = imp2 + (size_t)Bsz * Hh * Tmax;
    float* rhoArr  = massArr + Bsz * Hh;

    // weights -> bf16
    k_cvt_bf16<<<2048, 256, 0, stream>>>(qkv_w,  wqkv,  12L * 768 * 2304);
    k_cvt_bf16<<<2048, 256, 0, stream>>>(proj_w, wproj, 12L * 768 * 768);
    k_cvt_bf16<<<2048, 256, 0, stream>>>(fc1_w,  wf1,   12L * 768 * 3072);
    k_cvt_bf16<<<2048, 256, 0, stream>>>(fc2_w,  wf2,   12L * 3072 * 768);
    k_cvt_bf16<<<1024, 256, 0, stream>>>(patch_w, wpatch, 768L * 768);
    k_cvt_pad<<<1024, 256, 0, stream>>>(head_w, whead, 768, 1000, 1024);
    k_init<<<1, 1, 0, stream>>>(meta);

    // patch embed
    k_patchify<<<Bsz * NP, 256, 0, stream>>>(in_x, patches);
    {
        dim3 g(6, 98);
        k_gemm<<<g, 256, 0, stream>>>(patches, wpatch, patch_b, nullptr, pout, nullptr,
                                      768, 768, 768, 768, 768, 32, NP, NP, nullptr, 0, 0);
    }
    k_assemble<<<Bsz * 197, 256, 0, stream>>>(pout, cls_tok, pos_emb, xxA);

    float* bufs[2] = { xxA, xxB };
    dim3 gq(18, 99), gp(6, 99), gf1(24, 99), gf2(6, 99);
    dim3 gs(T16 * T16, Bsz * Hh), go(T16 * 4, Bsz * Hh);

    for (int i = 0; i < 12; ++i) {
        float* cur = bufs[i & 1];
        float* nxt = bufs[(i + 1) & 1];
        const float* n1w = norm1_w + i * Dm;  const float* n1b = norm1_b + i * Dm;
        const float* n2w = norm2_w + i * Dm;  const float* n2b = norm2_b + i * Dm;
        const unsigned short* wq = wqkv  + (size_t)i * 768 * 2304;
        const unsigned short* wp = wproj + (size_t)i * 768 * 768;
        const unsigned short* w1 = wf1   + (size_t)i * 768 * 3072;
        const unsigned short* w2 = wf2   + (size_t)i * 3072 * 768;
        const float* qb  = qkv_b  + i * 3 * Dm;
        const float* pb  = proj_b + i * Dm;
        const float* f1b = fc1_b  + i * FF;
        const float* f2b = fc2_b  + i * Dm;

        // ---- pass 1: cheap scores on un-pruned tokens ----
        k_layernorm<<<NTOT, 256, 0, stream>>>(cur, n1w, n1b, xn, meta, 1);
        k_gemm<<<gq, 256, 0, stream>>>(xn, wq, qb, nullptr, nullptr, qkvb,
                                       768, 2304, 2304, 768, 2304, 32, Tmax, 0, meta, 1, 0);
        k_split_qkv<<<NTOT, 256, 0, stream>>>(qkvb, qh, kh, vth, meta);
        k_cheap<<<Bsz * Hh, 256, 0, stream>>>(qh, kh, vth, imp2, massArr, rhoArr, meta);
        k_reduce_cheap<<<Tmax + 2, 32, 0, stream>>>(imp2, massArr, rhoArr, imp, fmeta);
        k_decide<<<1, 1, 0, stream>>>(meta);
        k_rank<<<1, 256, 0, stream>>>(imp, meta, map);
        k_gather<<<NTOT, 256, 0, stream>>>(cur, nxt, meta, map);
        k_commit<<<1, 1, 0, stream>>>(meta);

        // ---- pass 2: full block on pruned tokens ----
        k_layernorm<<<NTOT, 256, 0, stream>>>(nxt, n1w, n1b, xn, meta, 1);
        k_gemm<<<gq, 256, 0, stream>>>(xn, wq, qb, nullptr, nullptr, qkvb,
                                       768, 2304, 2304, 768, 2304, 32, Tmax, 0, meta, 1, 0);
        k_split_qkv<<<NTOT, 256, 0, stream>>>(qkvb, qh, kh, vth, meta);
        k_attn_scores<<<gs, 32, 0, stream>>>(qh, kh, scores, meta);
        k_softmax<<<Bsz * Hh * Tmax, 32, 0, stream>>>(scores, probs, meta);
        k_attn_out<<<go, 32, 0, stream>>>(probs, vth, obuf, meta);
        k_gemm<<<gp, 256, 0, stream>>>(obuf, wp, pb, nxt, nxt, nullptr,
                                       768, 768, 768, 768, 768, 32, Tmax, 0, meta, 1, 0);
        k_layernorm<<<NTOT, 256, 0, stream>>>(nxt, n2w, n2b, xn2, meta, 1);
        k_gemm<<<gf1, 256, 0, stream>>>(xn2, w1, f1b, nullptr, nullptr, hbuf,
                                        768, 3072, 3072, 768, 3072, 32, Tmax, 0, meta, 1, 1);
        k_gemm<<<gf2, 256, 0, stream>>>(hbuf, w2, f2b, nxt, nxt, nullptr,
                                        3072, 768, 768, 3072, 768, 32, Tmax, 0, meta, 1, 0);
    }

    // final norm on CLS rows + classification head
    k_ln_cls<<<Bsz, 256, 0, stream>>>(bufs[0], norm_w, norm_b, clsb);
    {
        dim3 g(8, 1);
        k_gemm<<<g, 256, 0, stream>>>(clsb, whead, head_b, nullptr, (float*)d_out, nullptr,
                                      768, 1024, 1000, 768, 1000, 32, 1, 1, nullptr, 0, 0);
    }
}
